// MarketGraphNet_70669391888468
// MI455X (gfx1250) — compile-verified, matrix operand order checked by reading, not run on hardware
//
#include <hip/hip_runtime.h>
#include <hip/hip_bf16.h>

// ---------------------------------------------------------------------------
// MarketGraphNet forward for MI455X (gfx1250, wave32, WMMA bf16 16x16x32).
// ---------------------------------------------------------------------------

typedef __bf16 bf16;
typedef __attribute__((ext_vector_type(8)))  __bf16 v8bf;
typedef __attribute__((ext_vector_type(16))) __bf16 v16bf;
typedef __attribute__((ext_vector_type(8)))  float  v8f;

#define NODES 20000
#define EDGES 320000

// ---------------------------------------------------------------- utilities

__global__ void fill_f32_k(float* __restrict__ p, float v, long long n) {
    long long i = (long long)blockIdx.x * blockDim.x + threadIdx.x;
    if (i < n) p[i] = v;
}

// W [K x Nout] row-major f32  ->  Wt [Nout x K] row-major bf16
__global__ void wcvt_t_k(const float* __restrict__ W, bf16* __restrict__ Wt,
                         int K, int Nout) {
    int idx = blockIdx.x * blockDim.x + threadIdx.x;
    if (idx >= K * Nout) return;
    int k = idx / Nout;
    int n = idx - k * Nout;
    Wt[(long long)n * K + k] = (bf16)W[idx];
}

__global__ void cvt_bf16_k(const float* __restrict__ x, bf16* __restrict__ y,
                           long long n) {
    long long i = (long long)blockIdx.x * blockDim.x + threadIdx.x;
    if (i < n) y[i] = (bf16)x[i];
}

__device__ inline void atomicMaxF(float* addr, float val) {
    unsigned int* ua = (unsigned int*)addr;
    unsigned int old = *ua;
    while (__uint_as_float(old) < val) {
        unsigned int assumed = old;
        old = atomicCAS(ua, assumed, __float_as_uint(val));
        if (old == assumed) break;
    }
}

// ------------------------------------------------------------- edge kernels
// thread = (edge, channel); channels coalesced. C = 1<<logC.

__global__ void edge_max_k(const float* __restrict__ X,
                           const long long* __restrict__ ei,
                           const float* __restrict__ t,
                           float* __restrict__ amax,
                           int logC, long long total) {
    long long idx = (long long)blockIdx.x * blockDim.x + threadIdx.x;
    if (idx >= total) return;
    long long e = idx >> logC;
    int c = (int)(idx & ((1 << logC) - 1));
    long long s = ei[e];
    long long d = ei[EDGES + e];
    float m = X[(s << logC) + c];
    atomicMaxF(&amax[(d << logC) + c], m * t[c]);
}

__global__ void edge_expsum_k(const float* __restrict__ X,
                              const long long* __restrict__ ei,
                              const float* __restrict__ t,
                              const float* __restrict__ amax,
                              float* __restrict__ denom,
                              float* __restrict__ numer,
                              int logC, long long total) {
    long long idx = (long long)blockIdx.x * blockDim.x + threadIdx.x;
    if (idx >= total) return;
    long long e = idx >> logC;
    int c = (int)(idx & ((1 << logC) - 1));
    long long s = ei[e];
    long long d = ei[EDGES + e];
    long long o = (d << logC) + c;
    float m = X[(s << logC) + c];
    float ev = __expf(m * t[c] - amax[o]);
    atomicAdd(&denom[o], ev);
    atomicAdd(&numer[o], m * ev);
}

// agg = numer/denom (0 for isolated nodes), emitted directly as bf16 GEMM-A.
__global__ void seg_div_cvt_k(const float* __restrict__ numer,
                              const float* __restrict__ denom,
                              bf16* __restrict__ aggbf, long long n) {
    long long i = (long long)blockIdx.x * blockDim.x + threadIdx.x;
    if (i >= n) return;
    float d = denom[i];
    aggbf[i] = (bf16)(d > 0.f ? numer[i] / d : 0.f);
}

// ------------------------------------------------------------ WMMA GEMM
// C[M x Nout] = A1@B1t' + A2@B2t' + bias  (Bt stored [Nout x K] row-major).
// Block: 8 waves, each wave one 16-row strip x 64 cols; grid.y tiles Nout/64.
// A/B fragment per 16-bit ISA layout: lane<16 -> K {0..7,16..23}+kb,
// lane>=16 -> K {8..15,24..31}+kb  ==> two b128 loads at koff and koff+16.

__device__ inline v16bf load_frag16(const bf16* __restrict__ p) {
    v8bf lo = *(const v8bf*)(p);
    v8bf hi = *(const v8bf*)(p + 16);
    v16bf r;
#pragma unroll
    for (int i = 0; i < 8; ++i) { r[i] = lo[i]; r[i + 8] = hi[i]; }
    return r;
}

__global__ __launch_bounds__(256)
void gemm2_wmma_k(const bf16* __restrict__ A1, const bf16* __restrict__ B1t, int K1,
                  const bf16* __restrict__ A2, const bf16* __restrict__ B2t, int K2,
                  const float* __restrict__ bias, float* __restrict__ C,
                  int M, int Nout) {
    const int lane = threadIdx.x & 31;
    const int wave = threadIdx.x >> 5;
    const int m0 = blockIdx.x * 128 + wave * 16;
    const int n0 = blockIdx.y * 64;
    const int lrow = lane & 15;
    const int koff = (lane >> 4) << 3;   // 0 or 8

    int arow = m0 + lrow;
    if (arow >= M) arow = M - 1;         // clamp; store is guarded

    v8f acc[4];
#pragma unroll
    for (int t = 0; t < 4; ++t) acc[t] = (v8f){0, 0, 0, 0, 0, 0, 0, 0};

    { // first product: A1 @ B1t'
        const bf16* ap = A1 + (long long)arow * K1 + koff;
        for (int kb = 0; kb < K1; kb += 32) {
            __builtin_prefetch(ap + kb + 64, 0, 3);
            v16bf a = load_frag16(ap + kb);
#pragma unroll
            for (int t = 0; t < 4; ++t) {
                const bf16* bp = B1t + (long long)(n0 + t * 16 + lrow) * K1 + koff + kb;
                v16bf b = load_frag16(bp);
                acc[t] = __builtin_amdgcn_wmma_f32_16x16x32_bf16(
                    false, a, false, b, (short)0, acc[t], false, false);
            }
        }
    }
    { // second product: A2 @ B2t'
        const bf16* ap = A2 + (long long)arow * K2 + koff;
        for (int kb = 0; kb < K2; kb += 32) {
            __builtin_prefetch(ap + kb + 64, 0, 3);
            v16bf a = load_frag16(ap + kb);
#pragma unroll
            for (int t = 0; t < 4; ++t) {
                const bf16* bp = B2t + (long long)(n0 + t * 16 + lrow) * K2 + koff + kb;
                v16bf b = load_frag16(bp);
                acc[t] = __builtin_amdgcn_wmma_f32_16x16x32_bf16(
                    false, a, false, b, (short)0, acc[t], false, false);
            }
        }
    }

    // C/D layout: lanes 0-15 rows r+0..7 (VGPR r -> M=r), lanes 16-31 M=r+8.
    const int mbase = m0 + ((lane >> 4) << 3);
#pragma unroll
    for (int t = 0; t < 4; ++t) {
        int n = n0 + t * 16 + lrow;
        float bv = bias[n];
#pragma unroll
        for (int r = 0; r < 8; ++r) {
            int m = mbase + r;
            if (m < M) C[(long long)m * Nout + n] = acc[t][r] + bv;
        }
    }
}

// ------------------------------------------------------------ graph LN
__global__ void reduce_sum_sq_k(const float* __restrict__ x, long long n,
                                float* __restrict__ stats) {
    __shared__ float s1[256], s2[256];
    long long stride = (long long)gridDim.x * 256;
    float a = 0.f, b = 0.f;
    for (long long i = (long long)blockIdx.x * 256 + threadIdx.x; i < n; i += stride) {
        float v = x[i];
        a += v; b += v * v;
    }
    s1[threadIdx.x] = a; s2[threadIdx.x] = b;
    __syncthreads();
    for (int o = 128; o > 0; o >>= 1) {
        if ((int)threadIdx.x < o) {
            s1[threadIdx.x] += s1[threadIdx.x + o];
            s2[threadIdx.x] += s2[threadIdx.x + o];
        }
        __syncthreads();
    }
    if (threadIdx.x == 0) {
        atomicAdd(&stats[0], s1[0]);
        atomicAdd(&stats[1], s2[0]);
    }
}

__global__ void graph_ln_relu_k(float* __restrict__ h, const float* __restrict__ stats,
                                const float* __restrict__ g, const float* __restrict__ beta,
                                int logC, long long total, float invTotal) {
    long long i = (long long)blockIdx.x * blockDim.x + threadIdx.x;
    if (i >= total) return;
    float mu = stats[0] * invTotal;
    float var = fmaxf(stats[1] * invTotal - mu * mu, 0.f);
    float scale = 1.f / (sqrtf(var) + 1e-5f);
    int c = (int)(i & ((1 << logC) - 1));
    float v = (h[i] - mu) * scale * g[c] + beta[c];
    h[i] = v > 0.f ? v : 0.f;
}

// ------------------------------------------------------------ head
__global__ void col_sum_k(const float* __restrict__ h, float* __restrict__ pool,
                          int n, int C) {
    int c = threadIdx.x;               // blockDim.x == C
    float s = 0.f;
    for (int r = blockIdx.x; r < n; r += gridDim.x)
        s += h[(long long)r * C + c];
    atomicAdd(&pool[c], s);
}

__global__ void final_head_k(const float* __restrict__ pool, const float* __restrict__ Wf,
                             const float* __restrict__ bfv, const float* __restrict__ gx,
                             const float* __restrict__ bx, float* __restrict__ out,
                             float invN) {
    __shared__ float q[64];
    int n = threadIdx.x;               // 64 threads
    float s = 0.f;
    for (int c = 0; c < 256; ++c)
        s += pool[c] * invN * Wf[c * 64 + n];
    s += bfv[n];
    q[n] = s;
    __syncthreads();
    float mu = 0.f;
    for (int i = 0; i < 64; ++i) mu += q[i];
    mu *= (1.f / 64.f);
    float var = 0.f;
    for (int i = 0; i < 64; ++i) { float d = q[i] - mu; var += d * d; }
    var *= (1.f / 64.f);
    float y = (s - mu) * rsqrtf(var + 1e-5f) * gx[n] + bx[n];
    out[n] = fmaxf(y, 0.f);
}

// ---------------------------------------------------------------- launcher

extern "C" void kernel_launch(void* const* d_in, const int* in_sizes, int n_in,
                              void* d_out, int out_size, void* d_ws, size_t ws_size,
                              hipStream_t stream) {
    (void)in_sizes; (void)n_in; (void)out_size; (void)ws_size;

    const float*     x    = (const float*)d_in[0];
    const long long* ei   = (const long long*)d_in[1];   // int64 [2,E]
    const float* t1  = (const float*)d_in[2];
    const float* Wl1 = (const float*)d_in[3];
    const float* bl1 = (const float*)d_in[4];
    const float* Wr1 = (const float*)d_in[5];
    const float* g1  = (const float*)d_in[6];
    const float* be1 = (const float*)d_in[7];
    const float* t2  = (const float*)d_in[8];
    const float* Wl2 = (const float*)d_in[9];
    const float* bl2 = (const float*)d_in[10];
    const float* Wr2 = (const float*)d_in[11];
    const float* g2  = (const float*)d_in[12];
    const float* be2 = (const float*)d_in[13];
    const float* Wf  = (const float*)d_in[14];
    const float* bfv = (const float*)d_in[15];
    const float* gx  = (const float*)d_in[16];
    const float* bx  = (const float*)d_in[17];
    float* out = (float*)d_out;

    // ---- workspace layout (256B aligned bump allocator) ----
    char* wp = (char*)d_ws;
    auto alloc = [&](size_t bytes) -> void* {
        void* p = wp;
        wp += (bytes + 255) & ~(size_t)255;
        return p;
    };
    const long long NC1 = (long long)NODES * 256;   // 5.12M
    const long long NC2 = (long long)NODES * 512;   // 10.24M
    float* amax  = (float*)alloc(NC2 * 4);
    float* numer = (float*)alloc(NC2 * 4);
    float* denom = (float*)alloc(NC2 * 4);
    float* h1    = (float*)alloc(NC2 * 4);          // [N,512]
    float* h2    = (float*)alloc(NC1 * 4);          // [N,256]
    bf16* aggbf  = (bf16*)alloc(NC2 * 2);
    bf16* xbf    = (bf16*)alloc(NC1 * 2);
    bf16* h1bf   = (bf16*)alloc(NC2 * 2);
    bf16* Wl1t   = (bf16*)alloc(256 * 512 * 2);     // [512 x 256]
    bf16* Wr1t   = (bf16*)alloc(256 * 512 * 2);
    bf16* Wl2t   = (bf16*)alloc(512 * 256 * 2);     // [256 x 512]
    bf16* Wr2t   = (bf16*)alloc(512 * 256 * 2);
    bf16* Wft    = (bf16*)alloc(256 * 64 * 2);      // [64 x 256]
    float* stats = (float*)alloc(8 * 4);
    float* pool  = (float*)alloc(256 * 4);

    const int TB = 256;
    auto blocks = [](long long n, int tb) { return (unsigned)((n + tb - 1) / tb); };
    const float NEG = -3.402823466e38f;

    // ---- weights: convert + transpose to bf16 [Nout x K] ----
    wcvt_t_k<<<blocks(256 * 512, TB), TB, 0, stream>>>(Wl1, Wl1t, 256, 512);
    wcvt_t_k<<<blocks(256 * 512, TB), TB, 0, stream>>>(Wr1, Wr1t, 256, 512);
    wcvt_t_k<<<blocks(512 * 256, TB), TB, 0, stream>>>(Wl2, Wl2t, 512, 256);
    wcvt_t_k<<<blocks(512 * 256, TB), TB, 0, stream>>>(Wr2, Wr2t, 512, 256);
    wcvt_t_k<<<blocks(256 * 64, TB), TB, 0, stream>>>(Wf, Wft, 256, 64);
    cvt_bf16_k<<<blocks(NC1, TB), TB, 0, stream>>>(x, xbf, NC1);

    // ================= Layer 1 (C = 256 -> 512) =================
    fill_f32_k<<<blocks(NC1, TB), TB, 0, stream>>>(amax, NEG, NC1);
    fill_f32_k<<<blocks(NC1, TB), TB, 0, stream>>>(numer, 0.f, NC1);
    fill_f32_k<<<blocks(NC1, TB), TB, 0, stream>>>(denom, 0.f, NC1);
    {
        long long tot = (long long)EDGES * 256;
        edge_max_k<<<blocks(tot, TB), TB, 0, stream>>>(x, ei, t1, amax, 8, tot);
        edge_expsum_k<<<blocks(tot, TB), TB, 0, stream>>>(x, ei, t1, amax, denom, numer, 8, tot);
    }
    seg_div_cvt_k<<<blocks(NC1, TB), TB, 0, stream>>>(numer, denom, aggbf, NC1);

    {   // h1 = agg@Wl1 + x@Wr1 + bl1   (M=20000, Nout=512, K=256+256)
        dim3 grid((NODES + 127) / 128, 512 / 64);
        gemm2_wmma_k<<<grid, 256, 0, stream>>>(aggbf, Wl1t, 256, xbf, Wr1t, 256,
                                               bl1, h1, NODES, 512);
    }
    fill_f32_k<<<1, 8, 0, stream>>>(stats, 0.f, 8);
    reduce_sum_sq_k<<<1024, 256, 0, stream>>>(h1, NC2, stats);
    graph_ln_relu_k<<<blocks(NC2, TB), TB, 0, stream>>>(h1, stats, g1, be1, 9, NC2,
                                                        1.f / (float)NC2);
    cvt_bf16_k<<<blocks(NC2, TB), TB, 0, stream>>>(h1, h1bf, NC2);

    // ================= Layer 2 (C = 512 -> 256) =================
    fill_f32_k<<<blocks(NC2, TB), TB, 0, stream>>>(amax, NEG, NC2);
    fill_f32_k<<<blocks(NC2, TB), TB, 0, stream>>>(numer, 0.f, NC2);
    fill_f32_k<<<blocks(NC2, TB), TB, 0, stream>>>(denom, 0.f, NC2);
    {
        long long tot = (long long)EDGES * 512;
        edge_max_k<<<blocks(tot, TB), TB, 0, stream>>>(h1, ei, t2, amax, 9, tot);
        edge_expsum_k<<<blocks(tot, TB), TB, 0, stream>>>(h1, ei, t2, amax, denom, numer, 9, tot);
    }
    seg_div_cvt_k<<<blocks(NC2, TB), TB, 0, stream>>>(numer, denom, aggbf, NC2);

    {   // h2 = agg@Wl2 + h1@Wr2 + bl2   (M=20000, Nout=256, K=512+512)
        dim3 grid((NODES + 127) / 128, 256 / 64);
        gemm2_wmma_k<<<grid, 256, 0, stream>>>(aggbf, Wl2t, 512, h1bf, Wr2t, 512,
                                               bl2, h2, NODES, 256);
    }
    fill_f32_k<<<1, 8, 0, stream>>>(stats, 0.f, 8);
    reduce_sum_sq_k<<<1024, 256, 0, stream>>>(h2, NC1, stats);
    graph_ln_relu_k<<<blocks(NC1, TB), TB, 0, stream>>>(h2, stats, g2, be2, 8, NC1,
                                                        1.f / (float)NC1);

    // ================= Head =================
    fill_f32_k<<<1, 256, 0, stream>>>(pool, 0.f, 256);
    col_sum_k<<<256, 256, 0, stream>>>(h2, pool, NODES, 256);
    final_head_k<<<1, 64, 0, stream>>>(pool, Wf, bfv, gx, bx, out, 1.f / (float)NODES);
}
